// OrdinalAwareSelfAttention_83348135346699
// MI455X (gfx1250) — matrix operand order repaired and checked by reading, not compiled, not run on hardware
//
#include <hip/hip_runtime.h>
#include <cstddef>

// ---- problem constants (match reference) ----
#define kE      1024   // EMBED_DIM
#define kH      16     // N_HEADS
#define kHD     64     // HEAD_DIM
#define kB      2
#define kS      2048
#define kNCATS  5

typedef __attribute__((ext_vector_type(16))) __bf16 v16bf;
typedef __attribute__((ext_vector_type(8)))  __bf16 v8bf;
typedef __attribute__((ext_vector_type(4)))  __bf16 v4bf;
typedef __attribute__((ext_vector_type(8)))  float  v8f;

struct bf2 { __bf16 hi, lo; };

__device__ __forceinline__ bf2 splitbf(float x) {
  bf2 r;
  r.hi = (__bf16)x;
  r.lo = (__bf16)(x - (float)r.hi);
  return r;
}

__device__ __forceinline__ v8f wmma_bf16(v16bf a, v16bf b, v8f c) {
  return __builtin_amdgcn_wmma_f32_16x16x32_bf16(false, a, false, b, (short)0, c,
                                                 false, false);
}

__device__ __forceinline__ v16bf cat8(v8bf a, v8bf b) {
  return __builtin_shufflevector(a, b, 0, 1, 2, 3, 4, 5, 6, 7,
                                       8, 9, 10, 11, 12, 13, 14, 15);
}

// ============================================================================
// Elementwise fp32 -> (bf16 hi, bf16 lo) split.  Done ONCE per operand so the
// GEMM/attention hot loops carry zero conversion VALU.  n multiple of 4.
// ============================================================================
__global__ void split_f32_kernel(const float* __restrict__ x,
                                 __bf16* __restrict__ hi,
                                 __bf16* __restrict__ lo, int n4) {
  const int i = blockIdx.x * blockDim.x + threadIdx.x;
  if (i >= n4) return;
  const int idx = i * 4;
  float4 v = *(const float4*)(x + idx);
  v4bf h, l;
  bf2 t0 = splitbf(v.x); h[0] = t0.hi; l[0] = t0.lo;
  bf2 t1 = splitbf(v.y); h[1] = t1.hi; l[1] = t1.lo;
  bf2 t2 = splitbf(v.z); h[2] = t2.hi; l[2] = t2.lo;
  bf2 t3 = splitbf(v.w); h[3] = t3.hi; l[3] = t3.lo;
  *(v4bf*)(hi + idx) = h;
  *(v4bf*)(lo + idx) = l;
}

// ============================================================================
// GEMM on pre-split operands:  Y = X @ W^T + bias.
//   X planes: [M,K] bf16 hi/lo.   W planes: [N,K] bf16 hi/lo.
// mode 0: Yf fp32 row-major [M,N]           (final output)
// mode 1: Yhi/Ylo planes, layout [B,H,S,D]  (Q, K)
// mode 2: Yhi/Ylo planes, layout [B,H,D,S]  (V, pre-transposed for P*V)
// One wave per 16(M) x 64(N) tile; 3 WMMA per 16x16x32 step (hi*hi+hi*lo+lo*hi).
// ============================================================================
__global__ void gemm_split_wmma(const __bf16* __restrict__ Xhi,
                                const __bf16* __restrict__ Xlo,
                                const __bf16* __restrict__ Whi,
                                const __bf16* __restrict__ Wlo,
                                const float*  __restrict__ bias,
                                float*  __restrict__ Yf,
                                __bf16* __restrict__ Yhi,
                                __bf16* __restrict__ Ylo,
                                int M, int Kdim, int N, int mode) {
  const int lane  = threadIdx.x;
  const int lmod  = lane & 15;
  const int lhalf = lane >> 4;
  const int n0 = blockIdx.x * 64;
  const int m0 = blockIdx.y * 16;

  v8f acc[4];
  #pragma unroll
  for (int nt = 0; nt < 4; ++nt) acc[nt] = (v8f){0,0,0,0,0,0,0,0};

  const __bf16* xh = Xhi + (size_t)(m0 + lmod) * Kdim;
  const __bf16* xl = Xlo + (size_t)(m0 + lmod) * Kdim;

  for (int k0 = 0; k0 < Kdim; k0 += 32) {
    if (k0 + 32 < Kdim) {                      // light prefetch of streaming A
      __builtin_prefetch(xh + k0 + 32, 0, 0);
      __builtin_prefetch(xl + k0 + 32, 0, 0);
    }
    // A fragment (16x32): e=0..15 -> k0+16*lhalf+e
    const int a0off = k0 + 16 * lhalf;
    v16bf ahi = cat8(*(const v8bf*)(xh + a0off), *(const v8bf*)(xh + a0off + 8));
    v16bf alo = cat8(*(const v8bf*)(xl + a0off), *(const v8bf*)(xl + a0off + 8));

    #pragma unroll
    for (int nt = 0; nt < 4; ++nt) {
      // B fragment (32x16) of W^T: lane col n, k = e + 16*lhalf (16 contiguous)
      const size_t woff = (size_t)(n0 + nt * 16 + lmod) * Kdim + k0 + 16 * lhalf;
      v16bf bhi = *(const v16bf*)(Whi + woff);
      v16bf blo = *(const v16bf*)(Wlo + woff);
      acc[nt] = wmma_bf16(ahi, bhi, acc[nt]);
      acc[nt] = wmma_bf16(ahi, blo, acc[nt]);
      acc[nt] = wmma_bf16(alo, bhi, acc[nt]);
    }
  }

  #pragma unroll
  for (int nt = 0; nt < 4; ++nt) {
    const int n  = n0 + nt * 16 + lmod;
    const float bv = bias[n];
    #pragma unroll
    for (int r = 0; r < 8; ++r) {
      const int m   = m0 + r + 8 * lhalf;
      const float v = acc[nt][r] + bv;
      if (mode == 0) {
        Yf[(size_t)m * N + n] = v;
      } else {
        const int bb = m / kS, ss = m % kS;
        const int hh = n / kHD, dd = n % kHD;
        const size_t idx = (mode == 1)
            ? ((((size_t)bb * kH + hh) * kS + ss) * kHD + dd)     // [B,H,S,D]
            : ((((size_t)bb * kH + hh) * kHD + dd) * kS + ss);    // [B,H,D,S]
        bf2 t = splitbf(v);
        Yhi[idx] = t.hi;
        Ylo[idx] = t.lo;
      }
    }
  }
}

// ============================================================================
// Flash attention, ordinal distance bias + causal mask, all-bf16-split inputs.
// One wave per (b*H+h, 16 query rows); streams 32-key chunks.
// Q/K planes [B,H,S,D]; V planes [B,H,D,S] (transposed -> contiguous B-frags).
// Emits attended as bf16 hi/lo planes [B,S,E] for the output projection.
// ============================================================================
__global__ void attn_wmma(const __bf16* __restrict__ Qhi, const __bf16* __restrict__ Qlo,
                          const __bf16* __restrict__ Khi, const __bf16* __restrict__ Klo,
                          const __bf16* __restrict__ Vthi, const __bf16* __restrict__ Vtlo,
                          const int*   __restrict__ resp,
                          const float* __restrict__ ord_embed,
                          __bf16* __restrict__ Atthi, __bf16* __restrict__ Attlo) {
  const int bh   = blockIdx.x;          // 0 .. B*H-1
  const int b    = bh / kH;
  const int h    = bh % kH;
  const int q0   = blockIdx.y * 16;
  const int lane = threadIdx.x;
  const int lmod  = lane & 15;
  const int lhalf = lane >> 4;

  __shared__ float ldsP[16][33];        // P transpose staging (padded)

  const __bf16* Qh = Qhi + (size_t)bh * kS * kHD;
  const __bf16* Ql = Qlo + (size_t)bh * kS * kHD;
  const __bf16* Kh = Khi + (size_t)bh * kS * kHD;
  const __bf16* Kl = Klo + (size_t)bh * kS * kHD;
  const __bf16* Vh = Vthi + (size_t)bh * kHD * kS;
  const __bf16* Vl = Vtlo + (size_t)bh * kHD * kS;
  const int*    rb = resp + b * kS;

  // softplus(ordinal_embed[0][h]); bias coef = 0.1 * w / (N_CATS-1)
  const float t    = ord_embed[h];
  const float w    = (t > 20.f) ? t : log1pf(expf(t));
  const float coef = 0.1f * 0.25f * w;
  const float scale = 0.125f;           // 1/sqrt(64)

  // Q A-fragments for both 32-wide d-chunks, kept in registers for whole loop
  v16bf qhi[2], qlo[2];
  {
    const __bf16* qrh = Qh + (size_t)(q0 + lmod) * kHD;
    const __bf16* qrl = Ql + (size_t)(q0 + lmod) * kHD;
    #pragma unroll
    for (int c = 0; c < 2; ++c) {
      const int off = c * 32 + 16 * lhalf;
      qhi[c] = cat8(*(const v8bf*)(qrh + off), *(const v8bf*)(qrh + off + 8));
      qlo[c] = cat8(*(const v8bf*)(qrl + off), *(const v8bf*)(qrl + off + 8));
    }
  }

  float rq[8];
  #pragma unroll
  for (int r = 0; r < 8; ++r) rq[r] = (float)rb[q0 + r + 8 * lhalf];

  v8f o[4];
  #pragma unroll
  for (int db = 0; db < 4; ++db) o[db] = (v8f){0,0,0,0,0,0,0,0};
  float mrun[8], lrun[8];
  #pragma unroll
  for (int r = 0; r < 8; ++r) { mrun[r] = -1e30f; lrun[r] = 0.f; }

  const int kend = q0 + 16;             // last visible key is q0+15
  for (int k0 = 0; k0 < kend; k0 += 32) {
    // ---- scores: two 16-key tiles, contraction over d in 2 chunks of 32 ----
    v8f sv[2];
    float rk[2];
    #pragma unroll
    for (int tt = 0; tt < 2; ++tt) {
      const int keyc = k0 + tt * 16 + lmod;
      rk[tt] = (float)rb[keyc];
      v8f a = (v8f){0,0,0,0,0,0,0,0};
      #pragma unroll
      for (int c = 0; c < 2; ++c) {
        const size_t koff = (size_t)keyc * kHD + c * 32 + 16 * lhalf;
        v16bf bhi = *(const v16bf*)(Kh + koff);
        v16bf blo = *(const v16bf*)(Kl + koff);
        a = wmma_bf16(qhi[c], bhi, a);
        a = wmma_bf16(qhi[c], blo, a);
        a = wmma_bf16(qlo[c], bhi, a);
      }
      sv[tt] = a;
    }

    // ---- scale, ordinal bias, causal mask ----
    #pragma unroll
    for (int tt = 0; tt < 2; ++tt) {
      const int keyg = k0 + tt * 16 + lmod;
      #pragma unroll
      for (int r = 0; r < 8; ++r) {
        const int row = q0 + r + 8 * lhalf;
        float s = sv[tt][r] * scale - coef * fabsf(rq[r] - rk[tt]);
        sv[tt][r] = (keyg <= row) ? s : -1e30f;
      }
    }

    // ---- online softmax (reductions across the 16-lane key groups) ----
    v8f p0 = sv[0], p1 = sv[1];
    #pragma unroll
    for (int r = 0; r < 8; ++r) {
      float v = fmaxf(p0[r], p1[r]);
      #pragma unroll
      for (int msk = 1; msk < 16; msk <<= 1) v = fmaxf(v, __shfl_xor(v, msk, 32));
      const float mnew  = fmaxf(mrun[r], v);
      const float alpha = expf(mrun[r] - mnew);
      const float e0 = expf(p0[r] - mnew);
      const float e1 = expf(p1[r] - mnew);
      p0[r] = e0; p1[r] = e1;
      float rs = e0 + e1;
      #pragma unroll
      for (int msk = 1; msk < 16; msk <<= 1) rs += __shfl_xor(rs, msk, 32);
      lrun[r] = lrun[r] * alpha + rs;
      mrun[r] = mnew;
      #pragma unroll
      for (int db = 0; db < 4; ++db) o[db][r] *= alpha;
    }

    // ---- transpose P through LDS (C-layout -> A-fragment layout) ----
    __syncthreads();
    #pragma unroll
    for (int r = 0; r < 8; ++r) {
      ldsP[r + 8 * lhalf][lmod]      = p0[r];
      ldsP[r + 8 * lhalf][16 + lmod] = p1[r];
    }
    __syncthreads();

    v16bf phi, plo;
    #pragma unroll
    for (int e = 0; e < 16; ++e) {
      int kl = e + 16 * lhalf;   // A-fragment key index
      bf2 tp = splitbf(ldsP[lmod][kl]);
      phi[e] = tp.hi; plo[e] = tp.lo;
    }

    // ---- O += P(16x32) x V(32x64); V^T planes give contiguous B-frags ----
    #pragma unroll
    for (int db = 0; db < 4; ++db) {
      const size_t voff = (size_t)(db * 16 + lmod) * kS + k0 + 16 * lhalf;
      v16bf vhi = *(const v16bf*)(Vh + voff);
      v16bf vlo = *(const v16bf*)(Vl + voff);
      o[db] = wmma_bf16(phi, vhi, o[db]);
      o[db] = wmma_bf16(phi, vlo, o[db]);
      o[db] = wmma_bf16(plo, vhi, o[db]);
    }
    __syncthreads();
  }

  // ---- finalize: divide by row sums, emit attended hi/lo planes [B,S,E] ----
  #pragma unroll
  for (int r = 0; r < 8; ++r) {
    const float inv = 1.0f / lrun[r];
    const int row = q0 + r + 8 * lhalf;
    #pragma unroll
    for (int db = 0; db < 4; ++db) {
      const size_t idx = ((size_t)b * kS + row) * kE + h * kHD + db * 16 + lmod;
      bf2 tv = splitbf(o[db][r] * inv);
      Atthi[idx] = tv.hi;
      Attlo[idx] = tv.lo;
    }
  }
}

// ============================================================================
extern "C" void kernel_launch(void* const* d_in, const int* in_sizes, int n_in,
                              void* d_out, int out_size, void* d_ws, size_t ws_size,
                              hipStream_t stream) {
  const float* query = (const float*)d_in[0];
  const float* key_  = (const float*)d_in[1];
  const float* value = (const float*)d_in[2];
  const int*   resp  = (const int*)  d_in[3];
  // d_in[4] = mask (tril) — causal mask applied analytically in-kernel.
  const float* Wq = (const float*)d_in[5];  const float* bq = (const float*)d_in[6];
  const float* Wk = (const float*)d_in[7];  const float* bk = (const float*)d_in[8];
  const float* Wv = (const float*)d_in[9];  const float* bv = (const float*)d_in[10];
  const float* Wo = (const float*)d_in[11]; const float* bo = (const float*)d_in[12];
  const float* ord = (const float*)d_in[13];

  float* out = (float*)d_out;

  const size_t NX = (size_t)kB * kS * kE;   // 4M elements (X / QKV / attended)
  const size_t NW = (size_t)kE * kE;        // 1M elements (weights)

  __bf16* p = (__bf16*)d_ws;
  __bf16* Qhi  = p;  p += NX;  __bf16* Qlo  = p;  p += NX;
  __bf16* Khi  = p;  p += NX;  __bf16* Klo  = p;  p += NX;
  __bf16* Vthi = p;  p += NX;  __bf16* Vtlo = p;  p += NX;
  __bf16* Athi = p;  p += NX;  __bf16* Atlo = p;  p += NX;
  __bf16* xqh  = p;  p += NX;  __bf16* xql  = p;  p += NX;
  __bf16* xkh  = p;  p += NX;  __bf16* xkl  = p;  p += NX;
  __bf16* xvh  = p;  p += NX;  __bf16* xvl  = p;  p += NX;
  __bf16* wqh  = p;  p += NW;  __bf16* wql  = p;  p += NW;
  __bf16* wkh  = p;  p += NW;  __bf16* wkl  = p;  p += NW;
  __bf16* wvh  = p;  p += NW;  __bf16* wvl  = p;  p += NW;
  __bf16* woh  = p;  p += NW;  __bf16* wol  = p;  p += NW;

  // ---- one-time operand splitting (bandwidth-bound, no redundant work) ----
  {
    const int tpb = 256;
    const int gx  = (int)(NX / 4 + tpb - 1) / tpb;
    const int gw  = (int)(NW / 4 + tpb - 1) / tpb;
    split_f32_kernel<<<gx, tpb, 0, stream>>>(query, xqh, xql, (int)(NX / 4));
    split_f32_kernel<<<gx, tpb, 0, stream>>>(key_,  xkh, xkl, (int)(NX / 4));
    split_f32_kernel<<<gx, tpb, 0, stream>>>(value, xvh, xvl, (int)(NX / 4));
    split_f32_kernel<<<gw, tpb, 0, stream>>>(Wq, wqh, wql, (int)(NW / 4));
    split_f32_kernel<<<gw, tpb, 0, stream>>>(Wk, wkh, wkl, (int)(NW / 4));
    split_f32_kernel<<<gw, tpb, 0, stream>>>(Wv, wvh, wvl, (int)(NW / 4));
    split_f32_kernel<<<gw, tpb, 0, stream>>>(Wo, woh, wol, (int)(NW / 4));
  }

  const int M = kB * kS;                    // 4096
  dim3 blk(32);
  dim3 ggemm(kE / 64, M / 16);              // (16, 256)

  // ---- projections: Q,K -> [B,H,S,D] planes; V -> [B,H,D,S] planes ----
  gemm_split_wmma<<<ggemm, blk, 0, stream>>>(xqh, xql, wqh, wql, bq,
                                             nullptr, Qhi, Qlo, M, kE, kE, 1);
  gemm_split_wmma<<<ggemm, blk, 0, stream>>>(xkh, xkl, wkh, wkl, bk,
                                             nullptr, Khi, Klo, M, kE, kE, 1);
  gemm_split_wmma<<<ggemm, blk, 0, stream>>>(xvh, xvl, wvh, wvl, bv,
                                             nullptr, Vthi, Vtlo, M, kE, kE, 2);

  // ---- flash attention ----
  dim3 gattn(kB * kH, kS / 16);             // (32, 128)
  attn_wmma<<<gattn, blk, 0, stream>>>(Qhi, Qlo, Khi, Klo, Vthi, Vtlo,
                                       resp, ord, Athi, Atlo);

  // ---- output projection -> fp32 d_out ----
  gemm_split_wmma<<<ggemm, blk, 0, stream>>>(Athi, Atlo, woh, wol, bo,
                                             out, nullptr, nullptr, M, kE, kE, 0);
}
